// MambaModel_1477468750111
// MI455X (gfx1250) — matrix operand order repaired and checked by reading, not compiled, not run on hardware
//
#include <hip/hip_runtime.h>
#include <hip/hip_bf16.h>

// ---------- model constants ----------
#define D_MODEL    1024
#define D_STATE    16
#define D_CONV     4
#define NUM_LAYERS 4
#define SEQ        512
#define BATCH      2
#define D_INNER    2048           // EXPAND * D_MODEL
#define DT_RANK    64             // D_MODEL / 16
#define MROWS      (BATCH * SEQ)  // 1024 flattened (b,l) rows
#define NDBL       (DT_RANK + 2 * D_STATE)  // 96

// ---------- WMMA vector types ----------
typedef __attribute__((ext_vector_type(16))) __bf16          v16bf;
typedef __attribute__((ext_vector_type(16))) unsigned short  v16u;
typedef __attribute__((ext_vector_type(8)))  unsigned short  v8u;
typedef __attribute__((ext_vector_type(8)))  float           v8f;

// f32 -> bf16 bits, round-to-nearest-even (used only in producers, not GEMM loop)
static __device__ __forceinline__ unsigned short f2bf(float f) {
    unsigned int u = __builtin_bit_cast(unsigned int, f);
    u += 0x7FFFu + ((u >> 16) & 1u);
    return (unsigned short)(u >> 16);
}

// ============================================================================
// bulk f32 -> bf16 conversion (weights, once per launch)
// ============================================================================
__global__ void cvt_f32_bf16(const float* __restrict__ in,
                             unsigned short* __restrict__ out, int n) {
    int i = blockIdx.x * blockDim.x + threadIdx.x;
    if (i < n) out[i] = f2bf(in[i]);
}

// ============================================================================
// WMMA GEMM, bf16 inputs, f32 accumulate:
//   C[m,n] = epilogue( sum_k A[m,k] * W[n,k] )
//   A bf16 row-major (lda), W bf16 row-major (ldb), outputs row-major (ldc)
//   wave -> 16(M) x 64(N); block = 128 thr = 4 waves -> 16 x 256 of C
//   mode 0: plain; mode 1: softplus(v + bias[n])
//   Cf (f32) and/or Cb (bf16) written if non-null.
// ============================================================================
__global__ void gemm_wmma_bf16(const unsigned short* __restrict__ A,
                               const unsigned short* __restrict__ W,
                               float* __restrict__ Cf,
                               unsigned short* __restrict__ Cb,
                               int N, int K, int lda, int ldb, int ldc,
                               const float* __restrict__ bias, int mode) {
    const int lane  = threadIdx.x & 31;
    const int wave  = threadIdx.x >> 5;
    const int m0    = blockIdx.x * 16;
    const int n0    = (blockIdx.y * 4 + wave) * 64;
    const int mloc  = lane & 15;
    const int khalf = lane >> 4;

    const unsigned short* arow = A + (size_t)(m0 + mloc) * lda + khalf * 16;

    v8f acc[4] = {};

    #pragma unroll 2
    for (int k0 = 0; k0 < K; k0 += 32) {
        // ---- A fragment 16x32: two contiguous 16B runs per lane ----
        v8u alo = *(const v8u*)(arow + k0);        // K = khalf*16 + e
        v8u ahi = *(const v8u*)(arow + k0 + 8);    // K = khalf*16 + 8 + e
        v16u au = __builtin_shufflevector(alo, ahi,
                    0,1,2,3,4,5,6,7,8,9,10,11,12,13,14,15);
        v16bf afrag = __builtin_bit_cast(v16bf, au);

        // ---- 4 B fragments 32x16: one contiguous 32B run per lane ----
        #pragma unroll
        for (int nf = 0; nf < 4; ++nf) {
            int n  = n0 + nf * 16 + mloc;
            int nc = (n < N) ? n : (N - 1);                 // clamp, EXEC stays full
            v16u bu = *(const v16u*)(W + (size_t)nc * ldb + k0 + khalf * 16);
            v16bf bfrag = __builtin_bit_cast(v16bf, bu);
            acc[nf] = __builtin_amdgcn_wmma_f32_16x16x32_bf16(
                false, afrag, false, bfrag, (short)0, acc[nf], false, false);
        }
    }

    // ---- store: lane n = lane&15; VGPR r -> m = r + 8*(lane>>4) ----
    #pragma unroll
    for (int nf = 0; nf < 4; ++nf) {
        int n = n0 + nf * 16 + mloc;
        if (n < N) {
            #pragma unroll
            for (int r = 0; r < 8; ++r) {
                int m   = m0 + r + 8 * khalf;
                float v = acc[nf][r];
                if (mode == 1) {
                    v += bias[n];
                    v = (v > 20.0f) ? v : logf(1.0f + __expf(v));   // softplus
                }
                if (Cf) Cf[(size_t)m * ldc + n] = v;
                if (Cb) Cb[(size_t)m * ldc + n] = f2bf(v);
            }
        }
    }
}

// ============================================================================
// x_bf[b,l,d] = bf16( pe[l,d] + condition[b]*to_cond_w[d] + to_cond_b[d] )
// ============================================================================
__global__ void init_x(const float* __restrict__ cond, const float* __restrict__ pe,
                       const float* __restrict__ tcw, const float* __restrict__ tcb,
                       unsigned short* __restrict__ x_bf) {
    int idx = blockIdx.x * blockDim.x + threadIdx.x;
    if (idx >= BATCH * SEQ * D_MODEL) return;
    int d = idx % D_MODEL;
    int l = (idx / D_MODEL) % SEQ;
    int b = idx / (D_MODEL * SEQ);
    x_bf[idx] = f2bf(pe[l * D_MODEL + d] + cond[b] * tcw[d] + tcb[d]);
}

// ============================================================================
// depthwise causal conv (width 4) + bias + silu; dual-write f32 + bf16
// ============================================================================
__global__ void conv_silu(const float* __restrict__ xz, const float* __restrict__ cw,
                          const float* __restrict__ cb,
                          float* __restrict__ xa_f, unsigned short* __restrict__ xa_bf) {
    int idx = blockIdx.x * blockDim.x + threadIdx.x;
    if (idx >= BATCH * SEQ * D_INNER) return;
    int c = idx % D_INNER;
    int l = (idx / D_INNER) % SEQ;
    int b = idx / (D_INNER * SEQ);
    float acc = cb[c];
    #pragma unroll
    for (int k = 0; k < D_CONV; ++k) {
        int ll = l + k - (D_CONV - 1);
        if (ll >= 0)
            acc += xz[((size_t)(b * SEQ + ll)) * (2 * D_INNER) + c] * cw[c * D_CONV + k];
    }
    acc = acc / (1.0f + __expf(-acc));                     // silu
    xa_f[idx]  = acc;
    xa_bf[idx] = f2bf(acc);
}

// ============================================================================
// selective scan: one thread per (b, d); writes bf16 y (only GEMM consumes it)
// ============================================================================
__global__ void selective_scan(const float* __restrict__ dt,
                               const float* __restrict__ x_dbl,
                               const float* __restrict__ xa_f,
                               const float* __restrict__ xz,   // z at col D_INNER
                               const float* __restrict__ A_log,
                               const float* __restrict__ Dskip,
                               unsigned short* __restrict__ y_bf) {
    int idx = blockIdx.x * blockDim.x + threadIdx.x;
    if (idx >= BATCH * D_INNER) return;
    int d = idx % D_INNER;
    int b = idx / D_INNER;

    float An[D_STATE], h[D_STATE];
    #pragma unroll
    for (int n = 0; n < D_STATE; ++n) {
        An[n] = -__expf(A_log[d * D_STATE + n]);
        h[n]  = 0.0f;
    }
    const float Dv = Dskip[d];

    for (int l = 0; l < SEQ; ++l) {
        size_t row = (size_t)(b * SEQ + l);
        float dtv = dt[row * D_INNER + d];
        float xav = xa_f[row * D_INNER + d];
        float zv  = xz[row * (2 * D_INNER) + D_INNER + d];
        const float* bd = x_dbl + row * NDBL;
        float acc = 0.0f;
        #pragma unroll
        for (int n = 0; n < D_STATE; ++n) {
            float da = __expf(dtv * An[n]);
            h[n] = da * h[n] + dtv * bd[DT_RANK + n] * xav;          // Bm
            acc += h[n] * bd[DT_RANK + D_STATE + n];                 // Cm
        }
        acc += xav * Dv;
        acc *= zv / (1.0f + __expf(-zv));                            // * silu(z)
        y_bf[row * D_INNER + d] = f2bf(acc);
    }
}

// ============================================================================
// host-side launch
// ============================================================================
extern "C" void kernel_launch(void* const* d_in, const int* in_sizes, int n_in,
                              void* d_out, int out_size, void* d_ws, size_t ws_size,
                              hipStream_t stream) {
    const float* condition = (const float*)d_in[0];
    const float* pe        = (const float*)d_in[1];
    const float* to_cond_w = (const float*)d_in[2];
    const float* to_cond_b = (const float*)d_in[3];
    const float* in_proj_w = (const float*)d_in[4];   // (4,4096,1024)
    const float* conv_w    = (const float*)d_in[5];   // (4,2048,4)
    const float* conv_b    = (const float*)d_in[6];   // (4,2048)
    const float* x_proj_w  = (const float*)d_in[7];   // (4,96,2048)
    const float* dt_proj_w = (const float*)d_in[8];   // (4,2048,64)
    const float* dt_proj_b = (const float*)d_in[9];   // (4,2048)
    const float* A_log     = (const float*)d_in[10];  // (4,2048,16)
    const float* D_skip    = (const float*)d_in[11];  // (4,2048)
    const float* out_proj_w= (const float*)d_in[12];  // (4,1024,2048)

    // ---- carve workspace ----
    char* p = (char*)d_ws;
    auto carve = [&](size_t bytes) { char* r = p; p += (bytes + 255) & ~(size_t)255; return r; };
    const int SZ_WIN  = NUM_LAYERS * 2 * D_INNER * D_MODEL;   // 16.8M
    const int SZ_WXP  = NUM_LAYERS * NDBL * D_INNER;          // 786K
    const int SZ_WDT  = NUM_LAYERS * D_INNER * DT_RANK;       // 524K
    const int SZ_WOUT = NUM_LAYERS * D_MODEL * D_INNER;       // 8.4M
    unsigned short* w_in_bf  = (unsigned short*)carve((size_t)SZ_WIN  * 2);
    unsigned short* w_xp_bf  = (unsigned short*)carve((size_t)SZ_WXP  * 2);
    unsigned short* w_dt_bf  = (unsigned short*)carve((size_t)SZ_WDT  * 2);
    unsigned short* w_out_bf = (unsigned short*)carve((size_t)SZ_WOUT * 2);
    unsigned short* x_bf     = (unsigned short*)carve((size_t)MROWS * D_MODEL * 2);
    float*          xz_f     = (float*)carve((size_t)MROWS * 2 * D_INNER * 4);
    float*          xa_f     = (float*)carve((size_t)MROWS * D_INNER * 4);
    unsigned short* xa_bf    = (unsigned short*)carve((size_t)MROWS * D_INNER * 2);
    float*          xdbl_f   = (float*)carve((size_t)MROWS * NDBL * 4);
    unsigned short* xdbl_bf  = (unsigned short*)carve((size_t)MROWS * NDBL * 2);
    float*          dt_f     = (float*)carve((size_t)MROWS * D_INNER * 4);
    unsigned short* y_bf     = (unsigned short*)carve((size_t)MROWS * D_INNER * 2);

    // ---- one-time weight conversion to bf16 (lives in L2 thereafter) ----
    cvt_f32_bf16<<<dim3((SZ_WIN  + 255) / 256), 256, 0, stream>>>(in_proj_w,  w_in_bf,  SZ_WIN);
    cvt_f32_bf16<<<dim3((SZ_WXP  + 255) / 256), 256, 0, stream>>>(x_proj_w,   w_xp_bf,  SZ_WXP);
    cvt_f32_bf16<<<dim3((SZ_WDT  + 255) / 256), 256, 0, stream>>>(dt_proj_w,  w_dt_bf,  SZ_WDT);
    cvt_f32_bf16<<<dim3((SZ_WOUT + 255) / 256), 256, 0, stream>>>(out_proj_w, w_out_bf, SZ_WOUT);

    // ---- x = pe + cond ----
    init_x<<<dim3((BATCH*SEQ*D_MODEL + 255) / 256), 256, 0, stream>>>(
        condition, pe, to_cond_w, to_cond_b, x_bf);

    for (int i = 0; i < NUM_LAYERS; ++i) {
        // xz = x @ in_w^T   (N=4096, K=1024) -> f32 only
        gemm_wmma_bf16<<<dim3(MROWS/16, 4096/256), 128, 0, stream>>>(
            x_bf, w_in_bf + (size_t)i * 2*D_INNER * D_MODEL, xz_f, nullptr,
            2*D_INNER, D_MODEL, D_MODEL, D_MODEL, 2*D_INNER, nullptr, 0);

        // depthwise conv + silu -> f32 (scan) + bf16 (GEMM A)
        conv_silu<<<dim3((BATCH*SEQ*D_INNER + 255) / 256), 256, 0, stream>>>(
            xz_f, conv_w + (size_t)i * D_INNER * D_CONV,
            conv_b + (size_t)i * D_INNER, xa_f, xa_bf);

        // x_dbl = xa @ xp_w^T  (N=96, K=2048) -> f32 (scan) + bf16 (dt GEMM A)
        gemm_wmma_bf16<<<dim3(MROWS/16, 1), 128, 0, stream>>>(
            xa_bf, w_xp_bf + (size_t)i * NDBL * D_INNER, xdbl_f, xdbl_bf,
            NDBL, D_INNER, D_INNER, D_INNER, NDBL, nullptr, 0);

        // dt = softplus(dt_lr @ dtp_w^T + b)  (N=2048, K=64, lda=96) -> f32
        gemm_wmma_bf16<<<dim3(MROWS/16, 2048/256), 128, 0, stream>>>(
            xdbl_bf, w_dt_bf + (size_t)i * D_INNER * DT_RANK, dt_f, nullptr,
            D_INNER, DT_RANK, NDBL, DT_RANK, D_INNER,
            dt_proj_b + (size_t)i * D_INNER, 1);

        // selective scan + gate -> bf16 y
        selective_scan<<<dim3((BATCH*D_INNER + 255) / 256), 256, 0, stream>>>(
            dt_f, xdbl_f, xa_f, xz_f,
            A_log + (size_t)i * D_INNER * D_STATE, D_skip + (size_t)i * D_INNER, y_bf);

        // out = y @ out_w^T  (N=1024, K=2048)
        if (i == NUM_LAYERS - 1) {
            gemm_wmma_bf16<<<dim3(MROWS/16, 1024/256), 128, 0, stream>>>(
                y_bf, w_out_bf + (size_t)i * D_MODEL * D_INNER, (float*)d_out, nullptr,
                D_MODEL, D_INNER, D_INNER, D_INNER, D_MODEL, nullptr, 0);
        } else {
            gemm_wmma_bf16<<<dim3(MROWS/16, 1024/256), 128, 0, stream>>>(
                y_bf, w_out_bf + (size_t)i * D_MODEL * D_INNER, nullptr, x_bf,
                D_MODEL, D_INNER, D_INNER, D_INNER, D_MODEL, nullptr, 0);
        }
    }
}
